// FastSelfAttention_86569360818688
// MI455X (gfx1250) — matrix operand order repaired and checked by reading, not compiled, not run on hardware
//
#include <hip/hip_runtime.h>
#include <stdint.h>

#define N_HEADS 16
#define HIDDEN  1024
#define HEAD    64
#define WINDOW  512
#define BB      4
#define SS      4096
#define ROWS    (BB * SS)            // 16384
#define INV_SQRT_DH 0.125f

typedef __attribute__((ext_vector_type(16))) __bf16 v16bf;
typedef __attribute__((ext_vector_type(8)))  __bf16 v8bf;
typedef __attribute__((ext_vector_type(8)))  float  v8f;

static __device__ __forceinline__ unsigned short f2bf(float f) {
    union { float f; unsigned u; } v; v.f = f;
    unsigned r = v.u + 0x7FFFu + ((v.u >> 16) & 1u);
    return (unsigned short)(r >> 16);
}

static __device__ __forceinline__ v16bf load_frag(const unsigned short* __restrict__ p0,
                                                  const unsigned short* __restrict__ p1) {
    v8bf lo = *(const v8bf*)p0;
    v8bf hi = *(const v8bf*)p1;
    v16bf r;
#pragma unroll
    for (int i = 0; i < 8; i++) { r[i] = lo[i]; r[i + 8] = hi[i]; }
    return r;
}

// ---------------- K0: convert weights f32 -> bf16 ----------------
__global__ void k_cvt_w(const float* __restrict__ w0, const float* __restrict__ w1,
                        const float* __restrict__ w2,
                        unsigned short* __restrict__ o0, unsigned short* __restrict__ o1,
                        unsigned short* __restrict__ o2) {
    int i = blockIdx.x * blockDim.x + threadIdx.x;
    if (i >= HIDDEN * HIDDEN) return;
    const float* src = (blockIdx.y == 0) ? w0 : ((blockIdx.y == 1) ? w1 : w2);
    unsigned short* dst = (blockIdx.y == 0) ? o0 : ((blockIdx.y == 1) ? o1 : o2);
    dst[(size_t)(i & (HIDDEN - 1)) * HIDDEN + (i >> 10)] = f2bf(src[i]);
}

// ---------------- K1: layernorm + additive-attention logits ----------------
__global__ __launch_bounds__(256) void k_ln(
    const float* __restrict__ x, const float* __restrict__ mask,
    const float* __restrict__ av, const float* __restrict__ lw, const float* __restrict__ lb,
    float* __restrict__ hs, unsigned short* __restrict__ hsb, float* __restrict__ aout) {
    __shared__ float s1[256], s2[256];
    __shared__ float lh[N_HEADS];
    const int row = blockIdx.x;
    const int tid = threadIdx.x;
    const float* xr = x + (size_t)row * HIDDEN;
    float v[4]; float sum = 0.f, sq = 0.f;
#pragma unroll
    for (int i = 0; i < 4; i++) { float t = xr[tid + i * 256]; v[i] = t; sum += t; sq += t * t; }
    s1[tid] = sum; s2[tid] = sq;
    __syncthreads();
    for (int o = 128; o > 0; o >>= 1) {
        if (tid < o) { s1[tid] += s1[tid + o]; s2[tid] += s2[tid + o]; }
        __syncthreads();
    }
    const float mean = s1[0] * (1.0f / HIDDEN);
    const float var  = s2[0] * (1.0f / HIDDEN) - mean * mean;
    const float rstd = rsqrtf(var + 1e-5f);
    if (tid < N_HEADS) lh[tid] = 0.f;
    __syncthreads();
    float* hr = hs + (size_t)row * HIDDEN;
    unsigned short* hb = hsb + (size_t)row * HIDDEN;
#pragma unroll
    for (int i = 0; i < 4; i++) {
        const int c = tid + i * 256;
        const float h = (v[i] - mean) * rstd * lw[c] + lb[c];
        hr[c] = h; hb[c] = f2bf(h);
        atomicAdd(&lh[c >> 6], h * av[c]);   // ds_add_f32, per-head dot
    }
    __syncthreads();
    if (tid < N_HEADS) {
        const float attn = lh[tid] * INV_SQRT_DH + mask[row];
        aout[(size_t)row * N_HEADS + tid] = __expf(attn);
    }
}

// ---------------- K2: block-local prefix sums (window-aligned) ----------------
__global__ void k_scan(const float* __restrict__ hs, const float* __restrict__ a,
                       float* __restrict__ ps, float* __restrict__ pz) {
    const int NBLK = SS / WINDOW;                 // 8
    const int NS = BB * HIDDEN * NBLK;            // 32768 channel-block scans for s
    int g = blockIdx.x * blockDim.x + threadIdx.x;
    if (g < NS) {
        const int c = g & (HIDDEN - 1);
        int rest = g >> 10;
        const int blk = rest & (NBLK - 1);
        const int b   = rest >> 3;
        const int h = c >> 6;
        const size_t r0 = (size_t)b * SS + (size_t)blk * WINDOW;
        float acc = 0.f;
        for (int j = 0; j < WINDOW; j++) {
            const size_t r = r0 + j;
            acc = fmaf(a[r * N_HEADS + h], hs[r * HIDDEN + c], acc);
            ps[r * HIDDEN + c] = acc;
        }
    } else if (g < NS + BB * N_HEADS * NBLK) {
        int g2 = g - NS;
        const int h = g2 & (N_HEADS - 1);
        int rest = g2 >> 4;
        const int blk = rest & (NBLK - 1);
        const int b   = rest >> 3;
        const size_t r0 = (size_t)b * SS + (size_t)blk * WINDOW;
        float acc = 0.f;
        for (int j = 0; j < WINDOW; j++) {
            const size_t r = r0 + j;
            acc += a[r * N_HEADS + h];
            pz[r * N_HEADS + h] = acc;
        }
    }
}

// ---------------- K3: windowed average -> attended (bf16) ----------------
__global__ void k_att(const float* __restrict__ ps, const float* __restrict__ pz,
                      unsigned short* __restrict__ attb) {
    const size_t i = (size_t)blockIdx.x * blockDim.x + threadIdx.x;  // over ROWS*HIDDEN
    const int c = (int)(i & (HIDDEN - 1));
    const size_t r = i >> 10;                 // global row
    const int t = (int)(r & (SS - 1));
    const int h = c >> 6;
    float sN = ps[i];
    float zN = pz[r * N_HEADS + h];
    if (t >= WINDOW) {
        const int j = t & (WINDOW - 1);
        const size_t rl = r - j - 1;          // last row of previous block (block total)
        const size_t rp = r - WINDOW;         // same offset in previous block
        sN += ps[rl * HIDDEN + c] - ps[rp * HIDDEN + c];
        zN += pz[rl * N_HEADS + h] - pz[rp * N_HEADS + h];
    }
    attb[i] = f2bf(sN / zN);
}

// ---------------- K4: bf16 WMMA GEMM, 32x64 tile/wave, 2-stage pipeline ----
__global__ __launch_bounds__(256) void k_gemm_bf16(
    const unsigned short* __restrict__ A, const unsigned short* __restrict__ W,
    float* __restrict__ C) {
    const int wave = threadIdx.x >> 5;
    const int lane = threadIdx.x & 31;
    const int m0 = (blockIdx.x * 8 + wave) * 32;   // 32 M-rows per wave
    const int n0 = blockIdx.y * 64;                // 64 N-cols per wave
    // A fragment lane mapping (16-bit A 16x32): lanes 0-15 -> K {0..7,16..23},
    // lanes 16-31 -> K {8..15,24..31}, row M = lane & 15.
    const int kloA = (lane < 16) ? 0 : 8;
    const unsigned short* a0 = A + (size_t)(m0 + (lane & 15)) * HIDDEN + kloA;
    const unsigned short* a1 = a0 + (size_t)16 * HIDDEN;
    // B fragment (W transposed to [N][K]): lane holds column N = lane & 15, same K order as A.
    const unsigned short* wb = W + (size_t)(n0 + (lane & 15)) * HIDDEN + kloA;

    v8f acc[8];
#pragma unroll
    for (int i = 0; i < 8; i++) acc[i] = (v8f){};

    v16bf A0[2], B0[4], A1[2], B1[4];

    // prologue: stage 0 <- k=0
    A0[0] = load_frag(a0, a0 + 16);
    A0[1] = load_frag(a1, a1 + 16);
#pragma unroll
    for (int nt = 0; nt < 4; nt++) B0[nt] = load_frag(wb + nt * 16 * HIDDEN, wb + nt * 16 * HIDDEN + 16);

    for (int k0 = 0; k0 < HIDDEN; k0 += 64) {
        // stage 1 loads <- k0+32 (always in range: k0 <= 960 -> 992)
        {
            const int kn = k0 + 32;
            __builtin_prefetch((const void*)(a0 + kn), 0, 0);
            const unsigned short* wk = wb + kn;
            A1[0] = load_frag(a0 + kn, a0 + kn + 16);
            A1[1] = load_frag(a1 + kn, a1 + kn + 16);
#pragma unroll
            for (int nt = 0; nt < 4; nt++)
                B1[nt] = load_frag(wk + nt * 16 * HIDDEN, wk + nt * 16 * HIDDEN + 16);
        }
        // compute with stage 0 (k0)
#pragma unroll
        for (int mi = 0; mi < 2; mi++)
#pragma unroll
            for (int nt = 0; nt < 4; nt++)
                acc[mi * 4 + nt] = __builtin_amdgcn_wmma_f32_16x16x32_bf16(
                    false, A0[mi], false, B0[nt], (short)0, acc[mi * 4 + nt], false, false);
        // stage 0 loads <- k0+64
        if (k0 + 64 < HIDDEN) {
            const int kn = k0 + 64;
            const unsigned short* wk = wb + kn;
            A0[0] = load_frag(a0 + kn, a0 + kn + 16);
            A0[1] = load_frag(a1 + kn, a1 + kn + 16);
#pragma unroll
            for (int nt = 0; nt < 4; nt++)
                B0[nt] = load_frag(wk + nt * 16 * HIDDEN, wk + nt * 16 * HIDDEN + 16);
        }
        // compute with stage 1 (k0+32)
#pragma unroll
        for (int mi = 0; mi < 2; mi++)
#pragma unroll
            for (int nt = 0; nt < 4; nt++)
                acc[mi * 4 + nt] = __builtin_amdgcn_wmma_f32_16x16x32_bf16(
                    false, A1[mi], false, B1[nt], (short)0, acc[mi * 4 + nt], false, false);
    }

    // C/D layout: VGPR r -> M = tile_m0 + (lane>=16 ? 8 : 0) + r ; N = n0 + (lane&15)
    const int cbase = n0 + (lane & 15);
#pragma unroll
    for (int mi = 0; mi < 2; mi++) {
        const int rbase = m0 + mi * 16 + ((lane >> 4) << 3);
#pragma unroll
        for (int nt = 0; nt < 4; nt++) {
            v8f av = acc[mi * 4 + nt];
#pragma unroll
            for (int r = 0; r < 8; r++)
                C[(size_t)(rbase + r) * HIDDEN + cbase + nt * 16] = av[r];
        }
    }
}

// ---------------- K5: per-(row,head) dot(q,k) * v ----------------
__global__ __launch_bounds__(256) void k_out(
    const float* __restrict__ q, const float* __restrict__ kk,
    const float* __restrict__ v, float* __restrict__ out) {
    const int wid  = blockIdx.x * (blockDim.x >> 5) + (threadIdx.x >> 5); // (row, head)
    const int lane = threadIdx.x & 31;
    const size_t base = (size_t)wid * HEAD + (size_t)lane * 2;
    const float2 q2 = *(const float2*)(q + base);
    const float2 k2 = *(const float2*)(kk + base);
    float p = q2.x * k2.x + q2.y * k2.y;
#pragma unroll
    for (int o = 16; o > 0; o >>= 1) p += __shfl_xor(p, o, 32);
    const float sim = p * INV_SQRT_DH;
    const float2 v2 = *(const float2*)(v + base);
    float2 o2; o2.x = sim * v2.x; o2.y = sim * v2.y;
    *(float2*)(out + base) = o2;
}

extern "C" void kernel_launch(void* const* d_in, const int* in_sizes, int n_in,
                              void* d_out, int out_size, void* d_ws, size_t ws_size,
                              hipStream_t stream) {
    (void)in_sizes; (void)n_in; (void)out_size; (void)ws_size;
    const float* x   = (const float*)d_in[0];
    const float* msk = (const float*)d_in[1];
    const float* av  = (const float*)d_in[2];
    const float* Wq  = (const float*)d_in[3];
    const float* Wk  = (const float*)d_in[4];
    const float* Wv  = (const float*)d_in[5];
    const float* lw  = (const float*)d_in[6];
    const float* lb  = (const float*)d_in[7];
    float* out = (float*)d_out;

    char* p = (char*)d_ws;
    auto alloc = [&](size_t bytes) -> char* {
        char* r = p; p += (bytes + 255) & ~(size_t)255; return r;
    };
    float*          hs   = (float*)alloc(sizeof(float) * (size_t)ROWS * HIDDEN);
    unsigned short* hsb  = (unsigned short*)alloc(2ull * ROWS * HIDDEN);
    float*          a    = (float*)alloc(sizeof(float) * (size_t)ROWS * N_HEADS);
    float*          ps   = (float*)alloc(sizeof(float) * (size_t)ROWS * HIDDEN);
    float*          pz   = (float*)alloc(sizeof(float) * (size_t)ROWS * N_HEADS);
    unsigned short* attb = (unsigned short*)alloc(2ull * ROWS * HIDDEN);
    unsigned short* wqb  = (unsigned short*)alloc(2ull * HIDDEN * HIDDEN);
    unsigned short* wkb  = (unsigned short*)alloc(2ull * HIDDEN * HIDDEN);
    unsigned short* wvb  = (unsigned short*)alloc(2ull * HIDDEN * HIDDEN);
    float*          qb   = (float*)alloc(sizeof(float) * (size_t)ROWS * HIDDEN);
    float*          kb   = (float*)alloc(sizeof(float) * (size_t)ROWS * HIDDEN);
    float*          vb   = (float*)alloc(sizeof(float) * (size_t)ROWS * HIDDEN);

    k_cvt_w<<<dim3(HIDDEN * HIDDEN / 256, 3), 256, 0, stream>>>(Wq, Wk, Wv, wqb, wkb, wvb);
    k_ln<<<ROWS, 256, 0, stream>>>(x, msk, av, lw, lb, hs, hsb, a);
    {
        const int nthreads = BB * HIDDEN * (SS / WINDOW) + BB * N_HEADS * (SS / WINDOW);
        k_scan<<<(nthreads + 255) / 256, 256, 0, stream>>>(hs, a, ps, pz);
    }
    k_att<<<(ROWS * (size_t)HIDDEN) / 256, 256, 0, stream>>>(ps, pz, attb);
    {
        dim3 gg(ROWS / 32 / 8, HIDDEN / 64);
        k_gemm_bf16<<<gg, 256, 0, stream>>>(hsb,  wqb, qb);
        k_gemm_bf16<<<gg, 256, 0, stream>>>(attb, wkb, kb);
        k_gemm_bf16<<<gg, 256, 0, stream>>>(attb, wvb, vb);
    }
    k_out<<<ROWS * N_HEADS / 8, 256, 0, stream>>>(qb, kb, vb, out);
}
